// PointerNetworkReadingOrder_87926570484374
// MI455X (gfx1250) — matrix operand order repaired and checked by reading, not compiled, not run on hardware
//
#include <hip/hip_runtime.h>
#include <hip/hip_bf16.h>
#include <math.h>

// ---------------------------------------------------------------------------
// MI455X / gfx1250: wave32, WMMA f32_16x16x32_f16 for all GEMMs,
// flash-attention style fused layers, bias recomputed on the fly.
// ---------------------------------------------------------------------------

typedef __attribute__((ext_vector_type(16))) _Float16 v16h;
typedef __attribute__((ext_vector_type(8)))  float    v8f;

#define N_ELEM   4096
#define D_MODEL  256
#define D_K      128
#define D_FF     2048
#define NUM_LAYERS 6
#define GEOM_TEMP 5.0f

#define WMMA16(a, b, c) \
  __builtin_amdgcn_wmma_f32_16x16x32_f16(false, (a), false, (b), (short)0, (c), false, false)

// ---- WMMA operand loaders (per CDNA5 ISA VGPR layouts, §7.12.2) -----------
// A: 16x32 f16. lane l: row m=l&15; K in {8*hi..8*hi+7} U {16+8*hi..+7}
__device__ __forceinline__ v16h load_a_tile(const _Float16* __restrict__ base, int ld) {
  const int l = threadIdx.x & 31;
  const int m = l & 15, hi = l >> 4;
  const _Float16* p = base + m * ld + 8 * hi;
  v16h a;
#pragma unroll
  for (int i = 0; i < 8; ++i) { a[i] = p[i]; a[8 + i] = p[16 + i]; }
  return a;
}

// B: 32x16 f16, B[k][n] = srcT[n][k] (srcT row-major, contiguous in k).
// lane l: col n=l&15; K in {8*hi..8*hi+7} U {16+8*hi..+7}
__device__ __forceinline__ v16h load_bT_tile(const _Float16* __restrict__ srcT, int ld) {
  const int l = threadIdx.x & 31;
  const int n = l & 15, hi = l >> 4;
  const _Float16* p = srcT + n * ld + 8 * hi;
  v16h b;
#pragma unroll
  for (int i = 0; i < 8; ++i) { b[i] = p[i]; b[8 + i] = p[16 + i]; }
  return b;
}

__device__ __forceinline__ float redmax16(float v) {
#pragma unroll
  for (int m = 1; m < 16; m <<= 1) v = fmaxf(v, __shfl_xor(v, m, 32));
  return v;
}
__device__ __forceinline__ float redsum16(float v) {
#pragma unroll
  for (int m = 1; m < 16; m <<= 1) v += __shfl_xor(v, m, 32);
  return v;
}
__device__ __forceinline__ float redsum32(float v) {
#pragma unroll
  for (int m = 1; m < 32; m <<= 1) v += __shfl_xor(v, m, 32);
  return v;
}
__device__ __forceinline__ float redmax32(float v) {
#pragma unroll
  for (int m = 1; m < 32; m <<= 1) v = fmaxf(v, __shfl_xor(v, m, 32));
  return v;
}

__device__ __forceinline__ v8f zero8() {
  v8f z = {0.f, 0.f, 0.f, 0.f, 0.f, 0.f, 0.f, 0.f};
  return z;
}

// geometric bias element (matches reference _geom_bias, normalized form)
__device__ __forceinline__ float bias_val(float cix, float ciy, float cjx, float cjy,
                                          int i, int j, float inv_maxd, float scale) {
  if (i == j) return 0.f;
  float dx = cix - cjx, dy = ciy - cjy;
  float dist = sqrtf(dx * dx + dy * dy + 1e-6f) * inv_maxd;
  float mult = (dx != 0.f ? 1.f : 0.f) + (dy != 0.f ? 1.f : 0.f);
  return mult * __expf(-3.f * dist) * scale;
}

// layer-norm of a 16x256 LDS tile -> global f32 out (one wave)
__device__ __forceinline__ void ln_store(float (*lsO)[D_MODEL], const float* __restrict__ g,
                                         const float* __restrict__ be, float* __restrict__ out,
                                         int r0, int lane) {
  for (int m = 0; m < 16; ++m) {
    float part = 0.f;
#pragma unroll
    for (int k = 0; k < 8; ++k) part += lsO[m][lane + 32 * k];
    float mu = redsum32(part) * (1.0f / 256.0f);
    float vv = 0.f;
#pragma unroll
    for (int k = 0; k < 8; ++k) { float d = lsO[m][lane + 32 * k] - mu; vv += d * d; }
    float rs = rsqrtf(redsum32(vv) * (1.0f / 256.0f) + 1e-5f);
#pragma unroll
    for (int k = 0; k < 8; ++k) {
      int c = lane + 32 * k;
      out[(size_t)(r0 + m) * D_MODEL + c] = (lsO[m][c] - mu) * rs * g[c] + be[c];
    }
  }
}

// ---------------------------------------------------------------------------
// small setup kernels
// ---------------------------------------------------------------------------
__global__ void k_init(float* scal) {
  if (threadIdx.x < 2) scal[threadIdx.x] = 0.f;
}

__global__ void k_centers(const float* __restrict__ bbox, float* cx, float* cy) {
  int i = blockIdx.x * blockDim.x + threadIdx.x;
  if (i < N_ELEM) {
    cx[i] = (bbox[i * 4 + 0] + bbox[i * 4 + 2]) * 0.5f;
    cy[i] = (bbox[i * 4 + 1] + bbox[i * 4 + 3]) * 0.5f;
  }
}

__global__ void k_maxd(const float* __restrict__ cx, const float* __restrict__ cy, float* scal) {
  float lm = 0.f;
  const long long total = (long long)N_ELEM * N_ELEM;
  for (long long idx = (long long)blockIdx.x * blockDim.x + threadIdx.x; idx < total;
       idx += (long long)gridDim.x * blockDim.x) {
    int i = (int)(idx >> 12), j = (int)(idx & 4095);
    float dx = cx[i] - cx[j], dy = cy[i] - cy[j];
    lm = fmaxf(lm, dx * dx + dy * dy);
  }
  lm = sqrtf(lm + 1e-6f);
  lm = redmax32(lm);
  if ((threadIdx.x & 31) == 0) atomicMax((unsigned int*)&scal[0], __float_as_uint(lm));
}

__global__ void k_maxb(const float* __restrict__ cx, const float* __restrict__ cy, float* scal) {
  const float inv_maxd = 1.f / (scal[0] + 1e-6f);
  float lm = 0.f;
  const long long total = (long long)N_ELEM * N_ELEM;
  for (long long idx = (long long)blockIdx.x * blockDim.x + threadIdx.x; idx < total;
       idx += (long long)gridDim.x * blockDim.x) {
    int i = (int)(idx >> 12), j = (int)(idx & 4095);
    if (i == j) continue;
    float dx = cx[i] - cx[j], dy = cy[i] - cy[j];
    float dist = sqrtf(dx * dx + dy * dy + 1e-6f) * inv_maxd;
    float mult = (dx != 0.f ? 1.f : 0.f) + (dy != 0.f ? 1.f : 0.f);
    lm = fmaxf(lm, mult * __expf(-3.f * dist));
  }
  lm = redmax32(lm);
  if ((threadIdx.x & 31) == 0) atomicMax((unsigned int*)&scal[1], __float_as_uint(lm));
}

// f32 [rows][cols] -> f16 transposed [cols][rows]
__global__ void k_convT(const float* __restrict__ src, _Float16* __restrict__ dst,
                        int rows, int cols) {
  int total = rows * cols;
  for (int idx = blockIdx.x * blockDim.x + threadIdx.x; idx < total;
       idx += gridDim.x * blockDim.x) {
    int r = idx / cols, c = idx - r * cols;
    dst[(size_t)c * rows + r] = (_Float16)src[idx];
  }
}

// X f32 -> Xh f16 row-major [4096][256] + XhT f16 [256][4096]
__global__ void k_prep(const float* __restrict__ X, _Float16* __restrict__ Xh,
                       _Float16* __restrict__ XhT) {
  int idx = blockIdx.x * blockDim.x + threadIdx.x;
  if (idx < N_ELEM * D_MODEL) {
    int r = idx >> 8, c = idx & 255;
    _Float16 h = (_Float16)X[idx];
    Xh[idx] = h;
    XhT[(size_t)c * N_ELEM + r] = h;
  }
}

// ---------------------------------------------------------------------------
// feature encoder: bbox MLP + cat embedding + sinusoidal(x0) -> fusion+relu
// one block (256 threads) per element
// ---------------------------------------------------------------------------
__global__ __launch_bounds__(256) void k_encode(
    const float* __restrict__ bbox, const int* __restrict__ cat,
    const float* __restrict__ w1, const float* __restrict__ bb1,
    const float* __restrict__ w2, const float* __restrict__ bb2,
    const float* __restrict__ cemb, const float* __restrict__ fw,
    const float* __restrict__ fb, float* __restrict__ Xout) {
  __shared__ float comb[384];
  __shared__ float t1[128];
  const int e = blockIdx.x, t = threadIdx.x;
  const float x0 = bbox[e * 4 + 0], y0 = bbox[e * 4 + 1];
  const float x1 = bbox[e * 4 + 2], y1 = bbox[e * 4 + 3];
  if (t < 128) {
    float s = bb1[t] + x0 * w1[0 * 128 + t] + y0 * w1[1 * 128 + t] +
              x1 * w1[2 * 128 + t] + y1 * w1[3 * 128 + t];
    t1[t] = fmaxf(s, 0.f);
  }
  __syncthreads();
  if (t < 128) {
    float s = bb2[t];
    for (int i = 0; i < 128; ++i) s += t1[i] * w2[i * 128 + t];
    comb[t] = fmaxf(s, 0.f);
    comb[128 + t] = cemb[cat[e] * 128 + t];
    int p = t >> 1;  // sinusoidal pair
    float ang = x0 * __expf(-9.210340372f * (float)(2 * p) * (1.f / 128.f));
    comb[256 + t] = (t & 1) ? __cosf(ang) : __sinf(ang);
  }
  __syncthreads();
  float s = fb[t];
  for (int i = 0; i < 384; ++i) s += comb[i] * fw[i * D_MODEL + t];
  Xout[(size_t)e * D_MODEL + t] = fmaxf(s, 0.f);
}

// ---------------------------------------------------------------------------
// fused attention layer: one wave per 16-row tile; flash softmax over 4096
// cols; bias recomputed from centers; epilogue residual + LN.
// ---------------------------------------------------------------------------
__global__ __launch_bounds__(32) void k_attn(
    const _Float16* __restrict__ Xh, const _Float16* __restrict__ XhT,
    const float* __restrict__ Xin, const float* __restrict__ cx,
    const float* __restrict__ cy, const float* __restrict__ scal,
    const float* __restrict__ g1, const float* __restrict__ be1,
    float* __restrict__ Xout) {
  const int lane = threadIdx.x & 31;
  const int n16 = lane & 15, hi = lane >> 4;
  const int r0 = blockIdx.x * 16;

  __shared__ _Float16 lsP[16][32];
  __shared__ float lsO[16][D_MODEL];

  // hoist A (16 rows x 256 K) in registers: 8 chunks of 16x32
  v16h aA[8];
#pragma unroll
  for (int k = 0; k < 8; ++k) aA[k] = load_a_tile(Xh + (size_t)r0 * D_MODEL + k * 32, D_MODEL);

  // row centers for this lane's rows (m = r + 8*hi)
  float cix[8], ciy[8];
#pragma unroll
  for (int r = 0; r < 8; ++r) {
    int i = r0 + r + 8 * hi;
    cix[r] = cx[i];
    ciy[r] = cy[i];
  }

  const float inv_maxd = 1.f / (scal[0] + 1e-6f);
  const float bscale = GEOM_TEMP / (scal[1] + 1e-6f);
  const float inv_dk = 0.0625f;  // 1/sqrt(256)

  v8f acc[16];
#pragma unroll
  for (int nt = 0; nt < 16; ++nt) acc[nt] = zero8();
  float rowmax[8], rowsum[8];
#pragma unroll
  for (int r = 0; r < 8; ++r) { rowmax[r] = -3.0e38f; rowsum[r] = 0.f; }

  for (int j0 = 0; j0 < N_ELEM; j0 += 32) {
    // scores for two 16-col tiles (K=256)
    v8f s0 = zero8(), s1 = zero8();
#pragma unroll
    for (int k = 0; k < 8; ++k) {
      v16h b0 = load_bT_tile(Xh + (size_t)j0 * D_MODEL + k * 32, D_MODEL);
      v16h b1 = load_bT_tile(Xh + (size_t)(j0 + 16) * D_MODEL + k * 32, D_MODEL);
      s0 = WMMA16(aA[k], b0, s0);
      s1 = WMMA16(aA[k], b1, s1);
    }
    const int ja = j0 + n16, jb = j0 + 16 + n16;
    const float cjx0 = cx[ja], cjy0 = cy[ja], cjx1 = cx[jb], cjy1 = cy[jb];

#pragma unroll
    for (int r = 0; r < 8; ++r) {
      const int i = r0 + r + 8 * hi;
      float v0 = s0[r] * inv_dk + bias_val(cix[r], ciy[r], cjx0, cjy0, i, ja, inv_maxd, bscale);
      float v1 = s1[r] * inv_dk + bias_val(cix[r], ciy[r], cjx1, cjy1, i, jb, inv_maxd, bscale);
      float t = redmax16(fmaxf(v0, v1));
      float mnew = fmaxf(rowmax[r], t);
      float corr = __expf(rowmax[r] - mnew);
      float p0 = __expf(v0 - mnew), p1 = __expf(v1 - mnew);
      rowsum[r] = rowsum[r] * corr + redsum16(p0 + p1);
      rowmax[r] = mnew;
#pragma unroll
      for (int nt = 0; nt < 16; ++nt) acc[nt][r] *= corr;
      lsP[r + 8 * hi][n16] = (_Float16)p0;
      lsP[r + 8 * hi][16 + n16] = (_Float16)p1;
    }
    __syncthreads();
    v16h ap = load_a_tile(&lsP[0][0], 32);
#pragma unroll
    for (int nt = 0; nt < 16; ++nt) {
      v16h b = load_bT_tile(XhT + (size_t)(nt * 16) * N_ELEM + j0, N_ELEM);
      acc[nt] = WMMA16(ap, b, acc[nt]);
    }
    __syncthreads();
  }

  // epilogue: normalize by rowsum, residual, layer-norm
#pragma unroll
  for (int nt = 0; nt < 16; ++nt) {
#pragma unroll
    for (int r = 0; r < 8; ++r) {
      int m = r + 8 * hi, c = nt * 16 + n16;
      lsO[m][c] = acc[nt][r] / rowsum[r] + Xin[(size_t)(r0 + m) * D_MODEL + c];
    }
  }
  __syncthreads();
  ln_store(lsO, g1, be1, Xout, r0, lane);
}

// ---------------------------------------------------------------------------
// fused FFN layer: H = relu(X@w1+b1) streamed in 32-col chunks, Y += H@w2,
// epilogue residual + LN.  w1h:[2048][256] (=w1^T), w2h:[256][2048] (=w2^T)
// ---------------------------------------------------------------------------
__global__ __launch_bounds__(32) void k_ffn(
    const _Float16* __restrict__ Xh, const float* __restrict__ Xin,
    const _Float16* __restrict__ w1h, const float* __restrict__ b1,
    const _Float16* __restrict__ w2h, const float* __restrict__ b2,
    const float* __restrict__ g2, const float* __restrict__ be2,
    float* __restrict__ Xout) {
  const int lane = threadIdx.x & 31;
  const int n16 = lane & 15, hi = lane >> 4;
  const int r0 = blockIdx.x * 16;

  __shared__ _Float16 lsP[16][32];
  __shared__ float lsO[16][D_MODEL];

  v16h aA[8];
#pragma unroll
  for (int k = 0; k < 8; ++k) aA[k] = load_a_tile(Xh + (size_t)r0 * D_MODEL + k * 32, D_MODEL);

  v8f acc[16];
#pragma unroll
  for (int nt = 0; nt < 16; ++nt) acc[nt] = zero8();

  for (int f0 = 0; f0 < D_FF; f0 += 32) {
    v8f h0 = zero8(), h1 = zero8();
#pragma unroll
    for (int k = 0; k < 8; ++k) {
      v16h b0 = load_bT_tile(w1h + (size_t)f0 * D_MODEL + k * 32, D_MODEL);
      v16h b1t = load_bT_tile(w1h + (size_t)(f0 + 16) * D_MODEL + k * 32, D_MODEL);
      h0 = WMMA16(aA[k], b0, h0);
      h1 = WMMA16(aA[k], b1t, h1);
    }
    const float bb0 = b1[f0 + n16], bb1 = b1[f0 + 16 + n16];
#pragma unroll
    for (int r = 0; r < 8; ++r) {
      lsP[r + 8 * hi][n16] = (_Float16)fmaxf(h0[r] + bb0, 0.f);
      lsP[r + 8 * hi][16 + n16] = (_Float16)fmaxf(h1[r] + bb1, 0.f);
    }
    __syncthreads();
    v16h ap = load_a_tile(&lsP[0][0], 32);
#pragma unroll
    for (int nt = 0; nt < 16; ++nt) {
      v16h b = load_bT_tile(w2h + (size_t)(nt * 16) * D_FF + f0, D_FF);
      acc[nt] = WMMA16(ap, b, acc[nt]);
    }
    __syncthreads();
  }

#pragma unroll
  for (int nt = 0; nt < 16; ++nt) {
#pragma unroll
    for (int r = 0; r < 8; ++r) {
      int m = r + 8 * hi, c = nt * 16 + n16;
      lsO[m][c] = acc[nt][r] + b2[c] + Xin[(size_t)(r0 + m) * D_MODEL + c];
    }
  }
  __syncthreads();
  ln_store(lsO, g2, be2, Xout, r0, lane);
}

// ---------------------------------------------------------------------------
// q/k projection: out f16 [4096][128] = X @ W + b ; wT is [128][256] (=W^T)
// ---------------------------------------------------------------------------
__global__ __launch_bounds__(32) void k_proj(const _Float16* __restrict__ Xh,
                                             const _Float16* __restrict__ wT,
                                             const float* __restrict__ bias,
                                             _Float16* __restrict__ out) {
  const int lane = threadIdx.x & 31;
  const int n16 = lane & 15, hi = lane >> 4;
  const int r0 = blockIdx.x * 16, c0 = blockIdx.y * 16;
  v8f c = zero8();
#pragma unroll
  for (int k = 0; k < 8; ++k) {
    v16h a = load_a_tile(Xh + (size_t)r0 * D_MODEL + k * 32, D_MODEL);
    v16h b = load_bT_tile(wT + (size_t)c0 * D_MODEL + k * 32, D_MODEL);
    c = WMMA16(a, b, c);
  }
  const int col = c0 + n16;
  const float bv = bias[col];
#pragma unroll
  for (int r = 0; r < 8; ++r)
    out[(size_t)(r0 + r + 8 * hi) * D_K + col] = (_Float16)(c[r] + bv);
}

// relation = (q @ k^T) / sqrt(128), f32 output
__global__ __launch_bounds__(32) void k_rel(const _Float16* __restrict__ qh,
                                            const _Float16* __restrict__ kh,
                                            float* __restrict__ out) {
  const int lane = threadIdx.x & 31;
  const int n16 = lane & 15, hi = lane >> 4;
  const int i0 = blockIdx.x * 16, j0 = blockIdx.y * 16;
  v8f c = zero8();
#pragma unroll
  for (int k = 0; k < 4; ++k) {
    v16h a = load_a_tile(qh + (size_t)i0 * D_K + k * 32, D_K);
    v16h b = load_bT_tile(kh + (size_t)j0 * D_K + k * 32, D_K);
    c = WMMA16(a, b, c);
  }
  const float s = 0.08838834764831845f;  // 1/sqrt(128)
#pragma unroll
  for (int r = 0; r < 8; ++r)
    out[(size_t)(i0 + r + 8 * hi) * N_ELEM + j0 + n16] = c[r] * s;
}

// ---------------------------------------------------------------------------
extern "C" void kernel_launch(void* const* d_in, const int* in_sizes, int n_in,
                              void* d_out, int out_size, void* d_ws, size_t ws_size,
                              hipStream_t stream) {
  (void)in_sizes; (void)n_in; (void)out_size; (void)ws_size;
  // setup_inputs() flattening order
  const float* bbox = (const float*)d_in[0];
  const int*   cat  = (const int*)d_in[1];
  const float* b1w  = (const float*)d_in[2];   // bbox1.w [4][128]
  const float* b1b  = (const float*)d_in[3];   // bbox1.b [128]
  const float* b2w  = (const float*)d_in[4];   // bbox2.w [128][128]
  const float* b2b  = (const float*)d_in[5];   // bbox2.b [128]
  const float* cemb = (const float*)d_in[6];   // cat_emb [8][128]
  const float* fw   = (const float*)d_in[7];   // fusion.w [384][256]
  const float* fb   = (const float*)d_in[8];   // fusion.b [256]
  const float* lw1  = (const float*)d_in[9];   // [6][256][2048]
  const float* lb1  = (const float*)d_in[10];  // [6][2048]
  const float* lw2  = (const float*)d_in[11];  // [6][2048][256]
  const float* lb2  = (const float*)d_in[12];  // [6][256]
  const float* lg1  = (const float*)d_in[13];
  const float* lbe1 = (const float*)d_in[14];
  const float* lg2  = (const float*)d_in[15];
  const float* lbe2 = (const float*)d_in[16];
  const float* qw   = (const float*)d_in[17];  // [256][128]
  const float* qb   = (const float*)d_in[18];
  const float* kw   = (const float*)d_in[19];
  const float* kb   = (const float*)d_in[20];

  // workspace carve-out (~27 MB)
  char* w = (char*)d_ws;
  auto alloc = [&](size_t bytes) {
    char* p = w;
    w += (bytes + 255) & ~(size_t)255;
    return p;
  };
  float*    cx   = (float*)alloc(N_ELEM * 4);
  float*    cy   = (float*)alloc(N_ELEM * 4);
  float*    scal = (float*)alloc(256);
  float*    Xa   = (float*)alloc((size_t)N_ELEM * D_MODEL * 4);
  float*    Xb   = (float*)alloc((size_t)N_ELEM * D_MODEL * 4);
  _Float16* Xh   = (_Float16*)alloc((size_t)N_ELEM * D_MODEL * 2);
  _Float16* XhT  = (_Float16*)alloc((size_t)D_MODEL * N_ELEM * 2);
  _Float16* w1h  = (_Float16*)alloc((size_t)NUM_LAYERS * D_FF * D_MODEL * 2);
  _Float16* w2h  = (_Float16*)alloc((size_t)NUM_LAYERS * D_MODEL * D_FF * 2);
  _Float16* qwh  = (_Float16*)alloc((size_t)D_K * D_MODEL * 2);
  _Float16* kwh  = (_Float16*)alloc((size_t)D_K * D_MODEL * 2);
  _Float16* qh   = (_Float16*)alloc((size_t)N_ELEM * D_K * 2);
  _Float16* kh   = (_Float16*)alloc((size_t)N_ELEM * D_K * 2);

  k_init<<<1, 32, 0, stream>>>(scal);
  k_centers<<<16, 256, 0, stream>>>(bbox, cx, cy);
  k_encode<<<N_ELEM, 256, 0, stream>>>(bbox, cat, b1w, b1b, b2w, b2b, cemb, fw, fb, Xa);
  k_maxd<<<2048, 256, 0, stream>>>(cx, cy, scal);
  k_maxb<<<2048, 256, 0, stream>>>(cx, cy, scal);

  for (int l = 0; l < NUM_LAYERS; ++l) {
    k_convT<<<1024, 256, 0, stream>>>(lw1 + (size_t)l * D_MODEL * D_FF,
                                      w1h + (size_t)l * D_FF * D_MODEL, D_MODEL, D_FF);
    k_convT<<<1024, 256, 0, stream>>>(lw2 + (size_t)l * D_FF * D_MODEL,
                                      w2h + (size_t)l * D_MODEL * D_FF, D_FF, D_MODEL);
  }
  k_convT<<<128, 256, 0, stream>>>(qw, qwh, D_MODEL, D_K);
  k_convT<<<128, 256, 0, stream>>>(kw, kwh, D_MODEL, D_K);

  for (int l = 0; l < NUM_LAYERS; ++l) {
    k_prep<<<4096, 256, 0, stream>>>(Xa, Xh, XhT);
    k_attn<<<N_ELEM / 16, 32, 0, stream>>>(Xh, XhT, Xa, cx, cy, scal,
                                           lg1 + l * D_MODEL, lbe1 + l * D_MODEL, Xb);
    k_prep<<<4096, 256, 0, stream>>>(Xb, Xh, XhT);
    k_ffn<<<N_ELEM / 16, 32, 0, stream>>>(Xh, Xb,
                                          w1h + (size_t)l * D_FF * D_MODEL, lb1 + l * D_FF,
                                          w2h + (size_t)l * D_MODEL * D_FF, lb2 + l * D_MODEL,
                                          lg2 + l * D_MODEL, lbe2 + l * D_MODEL, Xa);
  }

  k_prep<<<4096, 256, 0, stream>>>(Xa, Xh, XhT);
  k_proj<<<dim3(N_ELEM / 16, D_K / 16), 32, 0, stream>>>(Xh, qwh, qb, qh);
  k_proj<<<dim3(N_ELEM / 16, D_K / 16), 32, 0, stream>>>(Xh, kwh, kb, kh);
  k_rel<<<dim3(N_ELEM / 16, N_ELEM / 16), 32, 0, stream>>>(qh, kh, (float*)d_out);
}
